// LSTMAttentionDecoder_45655502356806
// MI455X (gfx1250) — compile-verified, matrix operand order checked by reading, not run on hardware
//
#include <hip/hip_runtime.h>
#include <hip/hip_bf16.h>
#include <math.h>

// ---------------- problem constants ----------------
#define Bn 128
#define Sn 2048
#define Hn 512
#define En 256
#define Vn 32000
#define SCHUNK 16
#define NCH (Sn / SCHUNK)   // 128

// ---------------- WMMA types -----------------------
typedef __attribute__((ext_vector_type(16))) __bf16       v16bf;
typedef __attribute__((ext_vector_type(8)))  float        v8f;
typedef __attribute__((ext_vector_type(8)))  unsigned int v8u;

union ABu { v8u u; v16bf b; };
__device__ __forceinline__ v16bf as_bf(v8u u) { ABu x; x.u = u; return x.b; }

// f32 -> bf16 round-to-nearest-even (scalar fallback)
__device__ __forceinline__ unsigned short f2bf(float f) {
  union { float f; unsigned int u; } v; v.f = f;
  unsigned int u = v.u;
  u += 0x7FFFu + ((u >> 16) & 1u);
  return (unsigned short)(u >> 16);
}
// packed f32x2 -> bf16x2 (hardware instruction when available)
__device__ __forceinline__ unsigned int pack_bf16(float x, float y) {
#if __has_builtin(__builtin_amdgcn_cvt_pk_bf16_f32)
  typedef __attribute__((ext_vector_type(2))) __bf16 v2bf;
  v2bf p = __builtin_amdgcn_cvt_pk_bf16_f32(x, y);
  union { v2bf v; unsigned int u; } c; c.v = p; return c.u;
#else
  return (unsigned int)f2bf(x) | ((unsigned int)f2bf(y) << 16);
#endif
}
__device__ __forceinline__ float sigmoidf_(float x) { return 1.0f / (1.0f + __expf(-x)); }

// ---- fragment-order addressing (ISA 7.12.2, 16-bit A/B layout, wave32) ----
// element (row, k) of a [rows][K] bf16 matrix -> flat ushort index such that
// lane L of (blk = row>>4, t = k>>5) reads its 8 fragment dwords contiguously.
__device__ __forceinline__ size_t fragpos(int row, int k, int K) {
  int blk = row >> 4, r = row & 15;
  int t = k >> 5, kl = k & 31;
  int hi = (kl >> 3) & 1;
  int i  = ((kl & 16) >> 2) | ((kl & 7) >> 1);
  int p  = k & 1;
  return ((((size_t)blk * (K >> 5) + t) * 32 + (hi * 16 + r)) * 8 + i) * 2 + p;
}

// ---------------- optional gfx1250 async LDS staging ----------------
#if __has_builtin(__builtin_amdgcn_global_load_async_to_lds_b128)
#define HAVE_ASYNC_LDS 1
typedef int v4i __attribute__((vector_size(16)));
typedef __attribute__((address_space(1))) v4i* g4p;   // global int4*
typedef __attribute__((address_space(3))) v4i* l4p;   // LDS int4*
__device__ __forceinline__ void async_b128(const void* g, void* l) {
  __builtin_amdgcn_global_load_async_to_lds_b128(
      (g4p)(unsigned long long)g,
      (l4p)(unsigned int)(unsigned long long)l, 0, 0);
}
__device__ __forceinline__ void async_wait0() {
#if __has_builtin(__builtin_amdgcn_s_wait_asynccnt)
  __builtin_amdgcn_s_wait_asynccnt(0);
#else
  asm volatile("s_wait_asynccnt 0x0" ::: "memory");
#endif
}
#endif

// ---------------- small utility kernels ----------------
// convert f32 [rows][K] -> bf16 fragment-order buffer
__global__ void conv_frag_kernel(const float* __restrict__ src,
                                 unsigned short* __restrict__ dst, int rows, int K) {
  int idx = blockIdx.x * blockDim.x + threadIdx.x;
  if (idx >= rows * K) return;
  int row = idx / K, k = idx - row * K;
  dst[fragpos(row, k, K)] = f2bf(src[idx]);
}

__global__ void embed_kernel(const int* __restrict__ tgt,
                             const float* __restrict__ emb,
                             unsigned short* __restrict__ xbf,      // frag [B][768]
                             unsigned short* __restrict__ cat2bf) { // frag [B][1280]
  int b = blockIdx.x;
  int t = tgt[b];  // tgt is (B,1)
  for (int e = threadIdx.x; e < En; e += blockDim.x) {
    unsigned short bv = f2bf(emb[(size_t)t * En + e]);
    xbf[fragpos(b, e, En + Hn)] = bv;                  // x = [embedded, context]
    cat2bf[fragpos(b, 2 * Hn + e, 2 * Hn + En)] = bv;  // [h1, context, embedded]
  }
}

// ---------------- generic M=128 WMMA GEMM ----------------
// OUT[m][n] = sum_k A1[m][k]*W1[n][k] + sum_k A2[m][k]*W2[n][k] + bias1[n] + bias2[n]
// A*: bf16 fragment-order [128][K]; W*: f32 row-major [N][K], staged+converted
// once per block into fragment-ordered LDS.  grid.x = N/16, block = 256.
__global__ __launch_bounds__(256)
void gemm_wmma_kernel(const unsigned int* __restrict__ A1f, const float* __restrict__ W1, int K1,
                      const unsigned int* __restrict__ A2f, const float* __restrict__ W2, int K2,
                      const float* __restrict__ bias1, const float* __restrict__ bias2,
                      float* __restrict__ OUT, int N) {
  __shared__ __attribute__((aligned(32))) unsigned int wt[16 * 1280 / 2];  // 40 KB, K<=1280
  int tid = threadIdx.x, wave = tid >> 5, lane = tid & 31;
  int r = lane & 15, hi = lane >> 4;
  int n0 = blockIdx.x * 16;
  v8f acc = {};

  for (int phase = 0; phase < 2; ++phase) {
    const unsigned int* Af = phase ? A2f : A1f;
    const float*        W  = phase ? W2  : W1;
    int                 K  = phase ? K2  : K1;
    if (Af == nullptr) continue;  // uniform across block

    // stage W tile (16 x K), convert once, fragment order
    int pairs = 16 * (K >> 1);
    for (int idx = tid; idx < pairs; idx += 256) {
      int row = idx / (K >> 1);
      int k   = (idx - row * (K >> 1)) * 2;
      float2 f = *(const float2*)(W + (size_t)(n0 + row) * K + k);
      wt[fragpos(row, k, K) >> 1] = pack_bf16(f.x, f.y);
    }
    __syncthreads();

    int tsteps = K >> 5;
    const unsigned int* Aw = Af + (size_t)wave * tsteps * 256;
    for (int t = 0; t < tsteps; ++t) {
      v8u a = *(const v8u*)(Aw + (size_t)(t * 32 + lane) * 8);
      v8u b = *(const v8u*)(wt + (size_t)(t * 32 + lane) * 8);
      acc = __builtin_amdgcn_wmma_f32_16x16x32_bf16(false, as_bf(a), false, as_bf(b),
                                                    (short)0, acc, false, false);
    }
    __syncthreads();
  }

  int n = n0 + r;
  float bias = 0.0f;
  if (bias1) bias += bias1[n];
  if (bias2) bias += bias2[n];
#pragma unroll
  for (int j = 0; j < 8; ++j) {
    int m = wave * 16 + j + 8 * hi;
    OUT[(size_t)m * N + n] = acc[j] + bias;
  }
}

// ---------------- fused attention pass (one read of encoder_outputs) -------
// grid = (NCH, B), block = 64 (2 waves; wave w owns n-tiles [w*16, w*16+16)).
__global__ __launch_bounds__(64)
void attn_partial_kernel(const float* __restrict__ enc,           // [B][S][H]
                         const unsigned short* __restrict__ Wencb,// frag [H][H] bf16
                         const float* __restrict__ dec_proj,      // [B][H]
                         const float* __restrict__ v_att,         // [H]
                         const unsigned char* __restrict__ src_mask, // [B][S]
                         float* __restrict__ raw_scores,          // [B][S]
                         float* __restrict__ chunk_m,             // [B][NCH]
                         float* __restrict__ chunk_l,             // [B][NCH]
                         float* __restrict__ ctx_part) {          // [B][NCH][H]
  __shared__ __attribute__((aligned(32))) float        encf[SCHUNK * Hn];      // 32 KB f32
  __shared__ __attribute__((aligned(32))) unsigned int encfr[SCHUNK * Hn / 2]; // 16 KB frag bf16
  __shared__ float scp[2][16];
  __shared__ float psarr[SCHUNK];
  __shared__ float sred[SCHUNK];

  int ch = blockIdx.x, b = blockIdx.y;
  int tid = threadIdx.x, wave = tid >> 5, lane = tid & 31;
  int r = lane & 15, hi = lane >> 4;
  int s0 = ch * SCHUNK;
  const float* encb = enc + ((size_t)b * Sn + s0) * Hn;

  // stage f32 chunk into LDS (async path when available)
#ifdef HAVE_ASYNC_LDS
  for (int q = tid; q < SCHUNK * Hn / 4; q += 64)
    async_b128(encb + q * 4, encf + q * 4);
  async_wait0();
  __syncthreads();
#else
  for (int q = tid; q < SCHUNK * Hn / 4; q += 64)
    *(float4*)(encf + q * 4) = *(const float4*)(encb + q * 4);
  __syncthreads();
#endif

  // convert to fragment-ordered bf16 (once per element)
  for (int q = tid; q < SCHUNK * Hn / 4; q += 64) {
    int e = q * 4, row = e >> 9, k = e & (Hn - 1);
    float4 f = *(const float4*)(encf + e);
    size_t dw = fragpos(row, k, Hn) >> 1;
    *(uint2*)(encfr + dw) = make_uint2(pack_bf16(f.x, f.y), pack_bf16(f.z, f.w));
  }
  __syncthreads();

  // proj[s,n] = sum_h enc[s,h]*W_enc[n,h]; fold tanh(.+dec_proj)*v into scores
  float sacc[8];
#pragma unroll
  for (int j = 0; j < 8; ++j) sacc[j] = 0.0f;

  const unsigned int* Wfr = (const unsigned int*)Wencb;
  for (int ntl = 0; ntl < 16; ++ntl) {
    int nt = wave * 16 + ntl;
    v8f acc = {};
    const unsigned int* Bf = Wfr + (size_t)nt * 4096;  // blk=nt, tsteps=16
    for (int t = 0; t < 16; ++t) {
      v8u a = *(const v8u*)(encfr + (size_t)(t * 32 + lane) * 8);
      v8u bb = *(const v8u*)(Bf + (size_t)(t * 32 + lane) * 8);
      acc = __builtin_amdgcn_wmma_f32_16x16x32_bf16(false, as_bf(a), false, as_bf(bb),
                                                    (short)0, acc, false, false);
    }
    int n = nt * 16 + r;
    float dp = dec_proj[(size_t)b * Hn + n];
    float vv = v_att[n];
#pragma unroll
    for (int j = 0; j < 8; ++j) sacc[j] += tanhf(acc[j] + dp) * vv;
  }

  // reduce over the 16 lanes of each half (m = j + 8*hi), per wave
#pragma unroll
  for (int j = 0; j < 8; ++j) {
    float x = sacc[j];
    for (int off = 1; off < 16; off <<= 1) x += __shfl_xor(x, off, 32);
    if (r == 0) scp[wave][j + 8 * hi] = x;
  }
  __syncthreads();

  // combine waves, mask, write raw scores
  if (tid < SCHUNK) {
    float s = scp[0][tid] + scp[1][tid];
    if (src_mask[(size_t)b * Sn + s0 + tid]) s = -1e10f;
    raw_scores[(size_t)b * Sn + s0 + tid] = s;
    sred[tid] = s;
  }
  __syncthreads();

  float mc = -INFINITY;
#pragma unroll
  for (int i = 0; i < SCHUNK; ++i) mc = fmaxf(mc, sred[i]);
  if (tid < SCHUNK) psarr[tid] = __expf(sred[tid] - mc);
  __syncthreads();
  float lc = 0.0f;
#pragma unroll
  for (int i = 0; i < SCHUNK; ++i) lc += psarr[i];
  if (tid == 0) {
    chunk_m[(size_t)b * NCH + ch] = mc;
    chunk_l[(size_t)b * NCH + ch] = lc;
  }

  // unnormalized context partial (f32 staging buffer, full precision)
  for (int h = tid; h < Hn; h += 64) {
    float a = 0.0f;
#pragma unroll
    for (int s = 0; s < SCHUNK; ++s) a += psarr[s] * encf[s * Hn + h];
    ctx_part[((size_t)b * NCH + ch) * Hn + h] = a;
  }
}

// ---------------- combine chunk partials -> attn + context ----------------
__global__ __launch_bounds__(256)
void attn_combine_kernel(const float* __restrict__ chunk_m, const float* __restrict__ chunk_l,
                         const float* __restrict__ ctx_part, const float* __restrict__ raw_scores,
                         float* __restrict__ attn_out,        // [B][S]
                         unsigned short* __restrict__ xbf,    // frag [B][768]
                         unsigned short* __restrict__ cat2bf) // frag [B][1280]
{
  int b = blockIdx.x, tid = threadIdx.x;
  __shared__ float em[NCH];
  __shared__ float Ms, Ls;
  if (tid == 0) {
    float m = -INFINITY;
    for (int c = 0; c < NCH; ++c) m = fmaxf(m, chunk_m[(size_t)b * NCH + c]);
    float l = 0.0f;
    for (int c = 0; c < NCH; ++c) {
      float e = __expf(chunk_m[(size_t)b * NCH + c] - m);
      em[c] = e;
      l += chunk_l[(size_t)b * NCH + c] * e;
    }
    Ms = m; Ls = l;
  }
  __syncthreads();
  float invL = 1.0f / Ls;

  for (int h = tid; h < Hn; h += blockDim.x) {
    float a = 0.0f;
    for (int c = 0; c < NCH; ++c) a += ctx_part[((size_t)b * NCH + c) * Hn + h] * em[c];
    a *= invL;
    unsigned short bv = f2bf(a);
    xbf[fragpos(b, En + h, En + Hn)] = bv;
    cat2bf[fragpos(b, Hn + h, 2 * Hn + En)] = bv;
  }
  float M = Ms;
  for (int s = tid; s < Sn; s += blockDim.x)
    attn_out[(size_t)b * Sn + s] = __expf(raw_scores[(size_t)b * Sn + s] - M) * invL;
}

// ---------------- LSTM elementwise activations ----------------
__global__ void lstm_act_kernel(const float* __restrict__ gates,  // [128][2048]
                                const float* __restrict__ c_prev, // [128][512]
                                float* __restrict__ h_out, float* __restrict__ c_out,
                                unsigned short* __restrict__ hbf, int hbfK, int hbfOff) {
  int idx = blockIdx.x * blockDim.x + threadIdx.x;
  if (idx >= Bn * Hn) return;
  int b = idx >> 9, j = idx & (Hn - 1);
  const float* g = gates + (size_t)b * (4 * Hn);
  float ig = sigmoidf_(g[j]);
  float fg = sigmoidf_(g[Hn + j]);
  float gg = tanhf(g[2 * Hn + j]);
  float og = sigmoidf_(g[3 * Hn + j]);
  float c = fg * c_prev[idx] + ig * gg;
  float h = og * tanhf(c);
  c_out[idx] = c;
  h_out[idx] = h;
  hbf[fragpos(b, hbfOff + j, hbfK)] = f2bf(h);
}

// ---------------- host-side launch ----------------
extern "C" void kernel_launch(void* const* d_in, const int* in_sizes, int n_in,
                              void* d_out, int out_size, void* d_ws, size_t ws_size,
                              hipStream_t stream) {
  const int*   tgt      = (const int*)  d_in[0];
  const float* hidden   = (const float*)d_in[1];   // [2][B][H]
  const float* cell     = (const float*)d_in[2];   // [2][B][H]
  const float* enc      = (const float*)d_in[3];   // [B][S][H]
  const unsigned char* src_mask = (const unsigned char*)d_in[4];
  const float* emb      = (const float*)d_in[5];
  const float* W_dec    = (const float*)d_in[6];
  const float* W_enc    = (const float*)d_in[7];
  const float* v_att    = (const float*)d_in[8];
  const float* W_ih0    = (const float*)d_in[9];
  const float* W_hh0    = (const float*)d_in[10];
  const float* b_ih0    = (const float*)d_in[11];
  const float* b_hh0    = (const float*)d_in[12];
  const float* W_ih1    = (const float*)d_in[13];
  const float* W_hh1    = (const float*)d_in[14];
  const float* b_ih1    = (const float*)d_in[15];
  const float* b_hh1    = (const float*)d_in[16];
  const float* W_out    = (const float*)d_in[17];
  const float* b_out    = (const float*)d_in[18];

  // output layout: logits | new_hidden(2,B,H) | new_cell(2,B,H) | attn(B,S)
  float* out = (float*)d_out;
  float* logits = out;
  float* h0_out = out + (size_t)Bn * Vn;
  float* h1_out = h0_out + (size_t)Bn * Hn;
  float* c0_out = h1_out + (size_t)Bn * Hn;
  float* c1_out = c0_out + (size_t)Bn * Hn;
  float* attn   = c1_out + (size_t)Bn * Hn;

  // workspace carve-out
  char* ws = (char*)d_ws;
  size_t off = 0;
  auto take = [&](size_t bytes) { char* p = ws + off; off = (off + bytes + 255) & ~(size_t)255; return p; };
  unsigned short* Wencb   = (unsigned short*)take((size_t)Hn * Hn * 2);          // frag
  unsigned short* dechb   = (unsigned short*)take((size_t)Bn * Hn * 2);          // hidden[1] frag
  unsigned short* h0prevb = (unsigned short*)take((size_t)Bn * Hn * 2);          // hidden[0] frag
  unsigned short* xbf     = (unsigned short*)take((size_t)Bn * (En + Hn) * 2);   // frag
  unsigned short* cat2bf  = (unsigned short*)take((size_t)Bn * (2 * Hn + En) * 2);
  unsigned short* h0bf    = (unsigned short*)take((size_t)Bn * Hn * 2);          // frag
  float* dec_proj   = (float*)take((size_t)Bn * Hn * 4);
  float* raw_scores = (float*)take((size_t)Bn * Sn * 4);
  float* chunk_m    = (float*)take((size_t)Bn * NCH * 4);
  float* chunk_l    = (float*)take((size_t)Bn * NCH * 4);
  float* ctx_part   = (float*)take((size_t)Bn * NCH * Hn * 4);
  float* gates      = (float*)take((size_t)Bn * 4 * Hn * 4);
  (void)ws_size; (void)in_sizes; (void)n_in; (void)out_size;

  // 1) fragment-order bf16 conversions of reused operands
  conv_frag_kernel<<<(Hn * Hn + 255) / 256, 256, 0, stream>>>(W_enc, Wencb, Hn, Hn);
  conv_frag_kernel<<<(Bn * Hn + 255) / 256, 256, 0, stream>>>(hidden + (size_t)Bn * Hn, dechb, Bn, Hn);
  conv_frag_kernel<<<(Bn * Hn + 255) / 256, 256, 0, stream>>>(hidden, h0prevb, Bn, Hn);

  // 2) embedding gather into x and final-concat buffers
  embed_kernel<<<Bn, 256, 0, stream>>>(tgt, emb, xbf, cat2bf);

  // 3) dec_proj = dec_h @ W_dec^T   (N = 512)
  gemm_wmma_kernel<<<Hn / 16, 256, 0, stream>>>((const unsigned int*)dechb, W_dec, Hn,
                                                nullptr, nullptr, 0,
                                                nullptr, nullptr, dec_proj, Hn);

  // 4) fused attention pass (single read of encoder_outputs)
  attn_partial_kernel<<<dim3(NCH, Bn), 64, 0, stream>>>(enc, Wencb, dec_proj, v_att,
                                                        src_mask, raw_scores,
                                                        chunk_m, chunk_l, ctx_part);

  // 5) combine -> attn output, context
  attn_combine_kernel<<<Bn, 256, 0, stream>>>(chunk_m, chunk_l, ctx_part, raw_scores,
                                              attn, xbf, cat2bf);

  // 6) LSTM layer 0 (N = 2048)
  gemm_wmma_kernel<<<(4 * Hn) / 16, 256, 0, stream>>>((const unsigned int*)xbf, W_ih0, En + Hn,
                                                      (const unsigned int*)h0prevb, W_hh0, Hn,
                                                      b_ih0, b_hh0, gates, 4 * Hn);
  lstm_act_kernel<<<(Bn * Hn + 255) / 256, 256, 0, stream>>>(gates, cell, h0_out, c0_out,
                                                             h0bf, Hn, 0);

  // 7) LSTM layer 1
  gemm_wmma_kernel<<<(4 * Hn) / 16, 256, 0, stream>>>((const unsigned int*)h0bf, W_ih1, Hn,
                                                      (const unsigned int*)dechb, W_hh1, Hn,
                                                      b_ih1, b_hh1, gates, 4 * Hn);
  lstm_act_kernel<<<(Bn * Hn + 255) / 256, 256, 0, stream>>>(gates, cell + (size_t)Bn * Hn,
                                                             h1_out, c1_out,
                                                             cat2bf, 2 * Hn + En, 0);

  // 8) logits = [h1, context, embedded] @ W_out^T + b_out   (N = 32000)
  gemm_wmma_kernel<<<Vn / 16, 256, 0, stream>>>((const unsigned int*)cat2bf, W_out, 2 * Hn + En,
                                                nullptr, nullptr, 0,
                                                b_out, nullptr, logits, Vn);
}
